// TemporalNeuralAttention_20083267076406
// MI455X (gfx1250) — matrix operand order repaired and checked by reading, not compiled, not run on hardware
//
#include <hip/hip_runtime.h>
#include <hip/hip_bf16.h>

typedef __attribute__((ext_vector_type(16))) _Float16 v16h;
typedef __attribute__((ext_vector_type(8)))  _Float16 v8h;
typedef __attribute__((ext_vector_type(8)))  float    v8f;

constexpr int TT = 512;   // sequence length
constexpr int DD = 512;   // model dim
constexpr int NN = 16;    // neurons
constexpr int HH = 8;     // heads
constexpr int HD = 64;    // head dim
constexpr int BN = 64;    // B * NN

// ---------------- WMMA helpers (16x16x32 f16 -> f32) ----------------

__device__ __forceinline__ v8f wmma_f16(v16h a, v16h b, v8f c) {
  return __builtin_amdgcn_wmma_f32_16x16x32_f16(
      /*neg_a=*/false, a, /*neg_b=*/false, b,
      /*c_mod=*/(short)0, c, /*reuse_a=*/false, /*reuse_b=*/false);
}

// A fragment: 16(M) x 32(K), row-major source with leading dim ld (f16 elems).
// lane: m = lane&15, kh = lane>>4. elems[0..15]=A[m][kh*16 + 0..15]
// (contiguous k order, matching load_b16).
__device__ __forceinline__ v16h load_a16(const _Float16* p, int ld, int lane) {
  int m = lane & 15, kh = lane >> 4;
  const _Float16* r = p + (size_t)m * ld + kh * 16;
  v8h lo = *(const v8h*)(r);
  v8h hi = *(const v8h*)(r + 8);
  v16h o;
#pragma unroll
  for (int e = 0; e < 8; e++) { o[e] = lo[e]; o[e + 8] = hi[e]; }
  return o;
}

// B fragment: 32(K) x 16(N), where column N=n of B is row n of the source
// matrix (i.e. source is [N][K] row-major, K contiguous).
// lane: n = lane&15, kh = lane>>4. elems = src[n][kh*16 .. kh*16+15].
__device__ __forceinline__ v16h load_b16(const _Float16* p, int ld, int lane) {
  int n = lane & 15, kh = lane >> 4;
  return *(const v16h*)(p + (size_t)n * ld + kh * 16);
}

// ---------------- Kernel 1: RMSNorm + RoPE -> f16 ----------------

__global__ void __launch_bounds__(256)
k_norm_rope(const float* __restrict__ x, const float* __restrict__ norm_w,
            _Float16* __restrict__ xr, _Float16* __restrict__ xn) {
  int t = blockIdx.x, bn = blockIdx.y;
  int b = bn >> 4, n = bn & 15;
  const float* row = x + ((((size_t)b * TT + t) * NN) + n) * DD;
  int i = threadIdx.x;                 // pair index 0..255
  float x0 = row[2 * i], x1 = row[2 * i + 1];
  __shared__ float red[256];
  red[i] = x0 * x0 + x1 * x1;
  __syncthreads();
  for (int s = 128; s > 0; s >>= 1) {
    if (i < s) red[i] += red[i + s];
    __syncthreads();
  }
  float rs = rsqrtf(red[0] * (1.0f / DD) + 1e-6f);
  float n0 = x0 * rs * norm_w[2 * i];
  float n1 = x1 * rs * norm_w[2 * i + 1];
  // inv_freq = 10000^(-2i/D) = exp(-(2i/D) * ln(10000))
  float inv = __expf(-((float)(2 * i) / (float)DD) * 9.210340371976184f);
  float ang = (float)t * inv;
  float s_, c_;
  __sincosf(ang, &s_, &c_);
  size_t o = ((size_t)bn * TT + t) * DD + 2 * i;
  xn[o] = (_Float16)n0;
  xn[o + 1] = (_Float16)n1;
  xr[o] = (_Float16)(n0 * c_ - n1 * s_);
  xr[o + 1] = (_Float16)(n0 * s_ + n1 * c_);
}

// ---------------- Kernel 2: weight convert f32 -> f16 ----------------

__global__ void __launch_bounds__(256)
k_cvt(const float* __restrict__ in_w, const float* __restrict__ out_w,
      _Float16* __restrict__ wIn, _Float16* __restrict__ wOut) {
  int idx = blockIdx.x * 256 + threadIdx.x;
  const int NIN = 3 * DD * DD;
  if (idx < NIN) wIn[idx] = (_Float16)in_w[idx];
  else           wOut[idx - NIN] = (_Float16)out_w[idx - NIN];
}

// ---------------- Kernel 3: QKV projection GEMM ----------------
// Per bn: M=T=512 rows, N=3D=1536 features, K=D=512.
// Block 256 thr = 8 waves (4 M x 2 N); wave tile 32(M) x 64(N).
// Features [0,512)=Q from xr, [512,1024)=K from xr, [1024,1536)=V from xn.
// V is written transposed: vT[bn][feat][t].

__global__ void __launch_bounds__(256)
k_qkv(const _Float16* __restrict__ xr, const _Float16* __restrict__ xn,
      const _Float16* __restrict__ wIn, const float* __restrict__ in_b,
      _Float16* __restrict__ qb, _Float16* __restrict__ kb,
      _Float16* __restrict__ vT) {
  int bn = blockIdx.z;
  int N0 = blockIdx.x * 128, M0 = blockIdx.y * 128;
  int lane = threadIdx.x & 31, w = threadIdx.x >> 5;
  int wm = w >> 1, wn = w & 1;
  const _Float16* A = (N0 < 2 * DD ? xr : xn) + (size_t)bn * TT * DD;
  int mBase = M0 + wm * 32, nBase = N0 + wn * 64;

  v8f acc[2][4] = {};
  for (int kk = 0; kk < DD; kk += 32) {
    v16h a0 = load_a16(A + (size_t)mBase * DD + kk, DD, lane);
    v16h a1 = load_a16(A + (size_t)(mBase + 16) * DD + kk, DD, lane);
#pragma unroll
    for (int j = 0; j < 4; j++) {
      v16h bf = load_b16(wIn + (size_t)(nBase + j * 16) * DD + kk, DD, lane);
      acc[0][j] = wmma_f16(a0, bf, acc[0][j]);
      acc[1][j] = wmma_f16(a1, bf, acc[1][j]);
    }
  }
  int col0 = lane & 15, half8 = (lane >> 4) * 8;
#pragma unroll
  for (int i = 0; i < 2; i++) {
#pragma unroll
    for (int j = 0; j < 4; j++) {
      int fTile = nBase + j * 16;
      int f = fTile + col0;
      float bias = in_b[f];
      int t0 = mBase + i * 16 + half8;
      if (fTile < 2 * DD) {           // Q or K, row-major
        _Float16* dst = (fTile < DD ? qb : kb) + (size_t)bn * TT * DD;
        int fc = f & (DD - 1);
#pragma unroll
        for (int r = 0; r < 8; r++)
          dst[(size_t)(t0 + r) * DD + fc] = (_Float16)(acc[i][j][r] + bias);
      } else {                        // V, transposed: contiguous in t
        int f2 = f - 2 * DD;
        _Float16* dst = vT + (size_t)bn * DD * TT + (size_t)f2 * TT + t0;
#pragma unroll
        for (int r = 0; r < 8; r++)
          dst[r] = (_Float16)(acc[i][j][r] + bias);
      }
    }
  }
}

// ---------------- Kernel 4: flash attention ----------------
// grid (T/64, H, BN), 128 thr = 4 waves; wave = 16 q-rows, 64-wide k tiles.

__global__ void __launch_bounds__(128)
k_attn(const _Float16* __restrict__ qb, const _Float16* __restrict__ kb,
       const _Float16* __restrict__ vT, const int* __restrict__ pad,
       _Float16* __restrict__ ao) {
  __shared__ __align__(32) _Float16 Pl[4][16][64];
  int qblk = blockIdx.x, h = blockIdx.y, bn = blockIdx.z;
  int lane = threadIdx.x & 31, w = threadIdx.x >> 5;
  int qBase = qblk * 64 + w * 16;
  int col0 = lane & 15, half8 = (lane >> 4) * 8;
  int rowTop = qBase + half8;

  const _Float16* Q = qb + (size_t)bn * TT * DD + (size_t)qBase * DD + h * HD;
  v16h aQ0 = load_a16(Q, DD, lane);       // K-dim 0..31 of head
  v16h aQ1 = load_a16(Q + 32, DD, lane);  // K-dim 32..63

  float rm[8], rsum[8];
  v8f o[4] = {};
#pragma unroll
  for (int r = 0; r < 8; r++) { rm[r] = -3.0e38f; rsum[r] = 0.0f; }
  bool padded = (pad[bn] == 0);

  for (int kt = 0; kt <= qblk; kt++) {
    // ---- S = (Q K^T) / sqrt(hd), 64x64 tile -> 4 accum tiles ----
    v8f s[4] = {};
#pragma unroll
    for (int nt = 0; nt < 4; nt++) {
      const _Float16* Kp =
          kb + (size_t)bn * TT * DD + (size_t)(kt * 64 + nt * 16) * DD + h * HD;
      v16h b0 = load_b16(Kp, DD, lane);
      v16h b1 = load_b16(Kp + 32, DD, lane);
      s[nt] = wmma_f16(aQ0, b0, s[nt]);
      s[nt] = wmma_f16(aQ1, b1, s[nt]);
    }
    // ---- scale + causal/pad mask (finfo.min semantics) ----
#pragma unroll
    for (int nt = 0; nt < 4; nt++) {
      int kc = kt * 64 + nt * 16 + col0;
#pragma unroll
      for (int r = 0; r < 8; r++) {
        float v = s[nt][r] * 0.125f;  // 1/sqrt(64)
        if (padded || kc > rowTop + r) v = -3.0e38f;
        s[nt][r] = v;
      }
    }
    // ---- online softmax: row max over 64 cols (16 lanes x 4 tiles) ----
    float nm[8], es[8];
#pragma unroll
    for (int r = 0; r < 8; r++) {
      float mx = fmaxf(fmaxf(s[0][r], s[1][r]), fmaxf(s[2][r], s[3][r]));
      mx = fmaxf(mx, __shfl_xor(mx, 1, 32));
      mx = fmaxf(mx, __shfl_xor(mx, 2, 32));
      mx = fmaxf(mx, __shfl_xor(mx, 4, 32));
      mx = fmaxf(mx, __shfl_xor(mx, 8, 32));
      nm[r] = fmaxf(rm[r], mx);
      es[r] = __expf(rm[r] - nm[r]);
      rm[r] = nm[r];
    }
    // ---- P = exp(S - m); stage into LDS (accum layout -> A layout) ----
#pragma unroll
    for (int nt = 0; nt < 4; nt++) {
#pragma unroll
      for (int r = 0; r < 8; r++) {
        float p = __expf(s[nt][r] - nm[r]);
        s[nt][r] = p;
        Pl[w][half8 + r][nt * 16 + col0] = (_Float16)p;
      }
    }
#pragma unroll
    for (int r = 0; r < 8; r++) {
      float ps = (s[0][r] + s[1][r]) + (s[2][r] + s[3][r]);
      ps += __shfl_xor(ps, 1, 32);
      ps += __shfl_xor(ps, 2, 32);
      ps += __shfl_xor(ps, 4, 32);
      ps += __shfl_xor(ps, 8, 32);
      rsum[r] = rsum[r] * es[r] + ps;
    }
    // ---- rescale O, then O += P V via WMMA ----
#pragma unroll
    for (int nt = 0; nt < 4; nt++)
#pragma unroll
      for (int r = 0; r < 8; r++) o[nt][r] *= es[r];

    v16h aP0 = load_a16(&Pl[w][0][0], 64, lane);
    v16h aP1 = load_a16(&Pl[w][0][32], 64, lane);
#pragma unroll
    for (int nt = 0; nt < 4; nt++) {
      const _Float16* Vp =
          vT + (size_t)bn * DD * TT + (size_t)(h * HD + nt * 16) * TT + kt * 64;
      v16h b0 = load_b16(Vp, TT, lane);
      v16h b1 = load_b16(Vp + 32, TT, lane);
      o[nt] = wmma_f16(aP0, b0, o[nt]);
      o[nt] = wmma_f16(aP1, b1, o[nt]);
    }
  }
  // ---- finalize: O /= rowsum, store f16 row-major ----
#pragma unroll
  for (int nt = 0; nt < 4; nt++) {
#pragma unroll
    for (int r = 0; r < 8; r++) {
      float v = o[nt][r] / rsum[r];
      ao[(size_t)bn * TT * DD + (size_t)(rowTop + r) * DD + h * HD + nt * 16 +
         col0] = (_Float16)v;
    }
  }
}

// ---------------- Kernel 5: output projection + bias + residual ----------------

__global__ void __launch_bounds__(256)
k_outproj(const _Float16* __restrict__ ao, const _Float16* __restrict__ wOut,
          const float* __restrict__ out_b, const float* __restrict__ x,
          float* __restrict__ out) {
  int bn = blockIdx.z;
  int b = bn >> 4, nn = bn & 15;
  int N0 = blockIdx.x * 128, M0 = blockIdx.y * 128;
  int lane = threadIdx.x & 31, w = threadIdx.x >> 5;
  int wm = w >> 1, wn = w & 1;
  const _Float16* A = ao + (size_t)bn * TT * DD;
  int mBase = M0 + wm * 32, nBase = N0 + wn * 64;

  v8f acc[2][4] = {};
  for (int kk = 0; kk < DD; kk += 32) {
    v16h a0 = load_a16(A + (size_t)mBase * DD + kk, DD, lane);
    v16h a1 = load_a16(A + (size_t)(mBase + 16) * DD + kk, DD, lane);
#pragma unroll
    for (int j = 0; j < 4; j++) {
      v16h bf = load_b16(wOut + (size_t)(nBase + j * 16) * DD + kk, DD, lane);
      acc[0][j] = wmma_f16(a0, bf, acc[0][j]);
      acc[1][j] = wmma_f16(a1, bf, acc[1][j]);
    }
  }
  int col0 = lane & 15, half8 = (lane >> 4) * 8;
#pragma unroll
  for (int i = 0; i < 2; i++) {
#pragma unroll
    for (int j = 0; j < 4; j++) {
      int f = nBase + j * 16 + col0;
      float bias = out_b[f];
      int t0 = mBase + i * 16 + half8;
#pragma unroll
      for (int r = 0; r < 8; r++) {
        size_t idx = ((((size_t)b * TT + (t0 + r)) * NN) + nn) * DD + f;
        out[idx] = acc[i][j][r] + bias + x[idx];
      }
    }
  }
}

// ---------------- launcher ----------------

extern "C" void kernel_launch(void* const* d_in, const int* in_sizes, int n_in,
                              void* d_out, int out_size, void* d_ws,
                              size_t ws_size, hipStream_t stream) {
  (void)in_sizes; (void)n_in; (void)out_size; (void)ws_size;
  const float* x      = (const float*)d_in[0];
  const int*   pad    = (const int*)d_in[1];
  const float* norm_w = (const float*)d_in[2];
  const float* in_w   = (const float*)d_in[3];
  const float* in_b   = (const float*)d_in[4];
  const float* out_w  = (const float*)d_in[5];
  const float* out_b  = (const float*)d_in[6];
  float* out = (float*)d_out;

  const size_t EL = (size_t)BN * TT * DD;  // 16.78M f16 elems per buffer
  _Float16* ws  = (_Float16*)d_ws;
  _Float16* xr  = ws;
  _Float16* xn  = xr + EL;
  _Float16* qb  = xn + EL;
  _Float16* kb  = qb + EL;
  _Float16* vT  = kb + EL;
  _Float16* ao  = vT + EL;
  _Float16* wIn = ao + EL;                  // 3*D*D
  _Float16* wOutH = wIn + (size_t)3 * DD * DD;

  k_cvt<<<dim3((4 * DD * DD) / 256), 256, 0, stream>>>(in_w, out_w, wIn, wOutH);
  k_norm_rope<<<dim3(TT, BN), 256, 0, stream>>>(x, norm_w, xr, xn);
  k_qkv<<<dim3(12, TT / 128, BN), 256, 0, stream>>>(xr, xn, wIn, in_b, qb, kb, vT);
  k_attn<<<dim3(TT / 64, HH, BN), 128, 0, stream>>>(qb, kb, vT, pad, ao);
  k_outproj<<<dim3(DD / 128, TT / 128, BN), 256, 0, stream>>>(ao, wOutH, out_b, x, out);
}
